// Block_31147102830605
// MI455X (gfx1250) — matrix operand order repaired and checked by reading, not compiled, not run on hardware
//
#include <hip/hip_runtime.h>
#include <hip/hip_bf16.h>
#include <math.h>

// ---------------- problem constants ----------------
#define BB 2
#define TT 512
#define CC 512
#define HH 8
#define DD 64
#define EE 8
#define FF 2048
#define TOPK 8

// ---------------- vector types ----------------
typedef float    f32x4 __attribute__((ext_vector_type(4)));
typedef float    v8f   __attribute__((ext_vector_type(8)));
typedef __bf16   v16bf __attribute__((ext_vector_type(16)));
typedef unsigned int u32x4 __attribute__((ext_vector_type(4)));

union Frag { v16bf v; u32x4 q[2]; };

__device__ __forceinline__ unsigned short f2bf(float f) {
    unsigned int u = __float_as_uint(f);
    unsigned int r = u + 0x7FFFu + ((u >> 16) & 1u);
    return (unsigned short)(r >> 16);
}

// ---------------- WMMA GEMM (double-buffered LDS, pipelined) ----------------
// C[M,N] = A[M,K] @ W[K,N] + bias  (+ epilogue variant)
// MODE 0: out = acc + bias
// MODE 1: out = acc + bias + extra[m*N+n]            (residual)
// MODE 2: out += (acc + bias) * extra[m*estride]     (row-scaled accumulate)
#define BM 64
#define BN 64
#define BK 32
#define APAD 40   // padded row stride (bf16 elems); 80B, 16B-aligned

template <int MODE>
__global__ __launch_bounds__(128)
void gemm_wmma(const float* __restrict__ A, const float* __restrict__ W,
               const float* __restrict__ bias, const float* __restrict__ extra,
               int estride, float* __restrict__ out, int M, int N, int K) {
    __shared__ __align__(16) unsigned short As[2][BM * APAD]; // 64 rows x 32 k
    __shared__ __align__(16) unsigned short Bs[2][BN * APAD]; // 64 cols x 32 k (transposed)

    const int tid  = threadIdx.x;
    const int wave = tid >> 5;
    const int lane = tid & 31;
    const int l    = lane & 15;
    const int hi   = lane >> 4;

    const int m0 = blockIdx.y * BM;
    const int n0 = blockIdx.x * BN;

    v8f acc[4] = {};

    // --- global tile staging helpers (per-thread: 4 float4 for A, 4 for B) ---
    auto loadA = [&](int k0, f32x4* r) {
        #pragma unroll
        for (int i = 0; i < 4; ++i) {
            int q  = tid + 128 * i;          // 0..511
            int rr = q >> 3;                 // 8 float4 per 32-wide row
            int c  = q & 7;
            r[i] = *(const f32x4*)(A + (size_t)(m0 + rr) * K + k0 + 4 * c);
        }
    };
    auto loadB = [&](int k0, f32x4* r) {
        #pragma unroll
        for (int i = 0; i < 4; ++i) {
            int q = tid + 128 * i;           // 0..511
            int k = q >> 4;                  // 16 float4 per 64-wide row
            int c = q & 15;
            r[i] = *(const f32x4*)(W + (size_t)(k0 + k) * N + n0 + 4 * c);
        }
    };
    auto stageA = [&](int buf, const f32x4* r) {
        #pragma unroll
        for (int i = 0; i < 4; ++i) {
            int q  = tid + 128 * i;
            int rr = q >> 3;
            int c  = q & 7;
            unsigned short* dst = &As[buf][rr * APAD + 4 * c];
            dst[0] = f2bf(r[i][0]); dst[1] = f2bf(r[i][1]);
            dst[2] = f2bf(r[i][2]); dst[3] = f2bf(r[i][3]);
        }
    };
    auto stageB = [&](int buf, const f32x4* r) {     // transpose to [n][k]
        #pragma unroll
        for (int i = 0; i < 4; ++i) {
            int q = tid + 128 * i;
            int k = q >> 4;
            int c = q & 15;
            #pragma unroll
            for (int j = 0; j < 4; ++j)
                Bs[buf][(4 * c + j) * APAD + k] = f2bf(r[i][j]);
        }
    };

    // --- prologue: stage first tile ---
    f32x4 ra[4], rb[4];
    loadA(0, ra); loadB(0, rb);
    stageA(0, ra); stageB(0, rb);

    int buf = 0;
    for (int k0 = 0; k0 < K; k0 += BK) {
        __syncthreads();                       // staged tile visible to all waves
        const bool has_next = (k0 + BK) < K;
        if (has_next) {                        // overlap global latency with WMMA
            loadA(k0 + BK, ra);
            loadB(k0 + BK, rb);
            if (k0 + 2 * BK < K)               // gfx1250 global_prefetch_b8
                __builtin_prefetch(W + (size_t)(k0 + 2 * BK) * N + n0, 0, 1);
        }

        // --- A fragment: lane<16 holds K{0..7,16..23}, lane>=16 K{8..15,24..31} ---
        Frag af;
        const u32x4* ap = (const u32x4*)(&As[buf][(wave * 16 + l) * APAD]);
        af.q[0] = ap[hi];
        af.q[1] = ap[2 + hi];

        // --- hoist all B fragments so the 4 WMMAs issue back-to-back ---
        Frag bf[4];
        #pragma unroll
        for (int j = 0; j < 4; ++j) {
            const u32x4* bp = (const u32x4*)(&Bs[buf][(16 * j + l) * APAD]);
            bf[j].q[0] = bp[hi];
            bf[j].q[1] = bp[2 + hi];
        }
        #pragma unroll
        for (int j = 0; j < 4; ++j) {
            acc[j] = __builtin_amdgcn_wmma_f32_16x16x32_bf16(
                false, af.v, false, bf[j].v, (short)0, acc[j], false, false);
        }

        if (has_next) {                        // stage into the idle buffer
            stageA(buf ^ 1, ra);
            stageB(buf ^ 1, rb);
        }
        buf ^= 1;
    }

    // --- epilogue: VGPR r -> M = r + 8*hi, N = lane&15 (per 16x16 tile) ---
    #pragma unroll
    for (int j = 0; j < 4; ++j) {
        #pragma unroll
        for (int r = 0; r < 8; ++r) {
            int m = m0 + wave * 16 + r + 8 * hi;
            int n = n0 + 16 * j + l;
            float val = acc[j][r] + bias[n];
            size_t o = (size_t)m * N + n;
            if (MODE == 0) out[o] = val;
            if (MODE == 1) out[o] = val + extra[o];
            if (MODE == 2) out[o] += val * extra[(size_t)m * estride];
        }
    }
}

// ---------------- RMSNorm ----------------
__global__ void rmsnorm_kernel(const float* __restrict__ x, const float* __restrict__ g,
                               const float* __restrict__ b, float* __restrict__ out) {
    __shared__ float red[256];
    const int row = blockIdx.x;
    const float* xr = x + (size_t)row * CC;
    float s = 0.f;
    for (int i = threadIdx.x; i < CC; i += 256) { float v = xr[i]; s += v * v; }
    red[threadIdx.x] = s; __syncthreads();
    for (int st = 128; st > 0; st >>= 1) {
        if (threadIdx.x < st) red[threadIdx.x] += red[threadIdx.x + st];
        __syncthreads();
    }
    float rms = sqrtf(red[0] / (float)CC + 1e-6f);
    for (int i = threadIdx.x; i < CC; i += 256)
        out[(size_t)row * CC + i] = xr[i] / rms * g[i] + b[i];
}

// ---------------- QK-norm + RoPE + head reshape ----------------
__global__ void qk_rope_kernel(const float* __restrict__ qkv, const float* __restrict__ alpha,
                               float* __restrict__ qh, float* __restrict__ kh,
                               float* __restrict__ vh) {
    const int h = blockIdx.x % HH;
    const int t = (blockIdx.x / HH) % TT;
    const int b = blockIdx.x / (HH * TT);
    const float* row = qkv + ((size_t)b * TT + t) * (3 * CC);
    __shared__ float sq[DD], sk[DD], red[DD];
    const int d = threadIdx.x;
    float qv = row[h * DD + d];
    float kv = row[CC + h * DD + d];
    float vv = row[2 * CC + h * DD + d];
    sq[d] = qv; sk[d] = kv;
    red[d] = qv * qv; __syncthreads();
    for (int st = 32; st > 0; st >>= 1) { if (d < st) red[d] += red[d + st]; __syncthreads(); }
    float nq = sqrtf(red[0]) + 1e-5f;
    __syncthreads();
    red[d] = kv * kv; __syncthreads();
    for (int st = 32; st > 0; st >>= 1) { if (d < st) red[d] += red[d + st]; __syncthreads(); }
    float nk = sqrtf(red[0]) + 1e-5f;
    __syncthreads();

    const float sclq = alpha[h] * 8.0f / nq;   // sqrt(D)=8
    const float sclk = 1.0f / nk;
    const size_t obase = (((size_t)b * HH + h) * TT + t) * DD;
    if (d < DD / 2) {
        float inv = __powf(10000.0f, -(float)d / 32.0f);
        float ang = (float)t * inv;
        float c = cosf(ang), s = sinf(ang);
        float q1 = sq[2 * d] * sclq, q2 = sq[2 * d + 1] * sclq;
        qh[obase + 2 * d]     = q1 * c - q2 * s;
        qh[obase + 2 * d + 1] = q1 * s + q2 * c;
        float k1 = sk[2 * d] * sclk, k2 = sk[2 * d + 1] * sclk;
        kh[obase + 2 * d]     = k1 * c - k2 * s;
        kh[obase + 2 * d + 1] = k1 * s + k2 * c;
    }
    vh[obase + d] = vv;
}

// ---------------- causal attention, one (b,h,t) row per block ----------------
__global__ __launch_bounds__(128)
void attention_kernel(const float* __restrict__ qh, const float* __restrict__ kh,
                      const float* __restrict__ vh, float* __restrict__ y) {
    __shared__ float qrow[DD];
    __shared__ float sc[TT];
    __shared__ float red[128];
    const int t = blockIdx.x % TT;
    const int h = (blockIdx.x / TT) % HH;
    const int b = blockIdx.x / (TT * HH);
    const size_t base = ((size_t)b * HH + h) * TT * DD;
    const int tid = threadIdx.x;
    if (tid < DD) qrow[tid] = qh[base + (size_t)t * DD + tid];
    __syncthreads();

    const int nk = t + 1;
    float lmax = -3.4e38f;
    for (int k = tid; k < nk; k += 128) {
        const float* kr = kh + base + (size_t)k * DD;
        float s = 0.f;
        #pragma unroll 8
        for (int i = 0; i < DD; ++i) s += qrow[i] * kr[i];
        s *= 0.125f;                      // 1/sqrt(D)
        sc[k] = s;
        lmax = fmaxf(lmax, s);
    }
    red[tid] = lmax; __syncthreads();
    for (int st = 64; st > 0; st >>= 1) {
        if (tid < st) red[tid] = fmaxf(red[tid], red[tid + st]);
        __syncthreads();
    }
    float mx = red[0]; __syncthreads();

    float lsum = 0.f;
    for (int k = tid; k < nk; k += 128) {
        float e = __expf(sc[k] - mx);
        sc[k] = e; lsum += e;
    }
    red[tid] = lsum; __syncthreads();
    for (int st = 64; st > 0; st >>= 1) {
        if (tid < st) red[tid] += red[tid + st];
        __syncthreads();
    }
    float S = red[0]; __syncthreads();

    const int d = tid & 63, part = tid >> 6;
    float acc = 0.f;
    for (int k = part; k < nk; k += 2) acc += sc[k] * vh[base + (size_t)k * DD + d];
    red[tid] = acc; __syncthreads();
    if (tid < DD)
        y[((size_t)b * TT + t) * CC + h * DD + tid] = (red[tid] + red[tid + 64]) / S;
}

// ---------------- router: logits, softmax, full sort (TOPK==E) ----------------
__global__ void router_kernel(const float* __restrict__ x1, const float* __restrict__ Wr,
                              const float* __restrict__ br, float* __restrict__ logits,
                              float* __restrict__ topi, float* __restrict__ topp,
                              float* __restrict__ cw) {
    __shared__ float lg[EE];
    const int bt = blockIdx.x;
    const float* xr = x1 + (size_t)bt * CC;
    const int e = threadIdx.x;
    if (e < EE) {
        float acc = br[e];
        for (int c = 0; c < CC; ++c) acc += xr[c] * Wr[c * EE + e];
        lg[e] = acc;
        logits[(size_t)bt * EE + e] = acc;
    }
    __syncthreads();
    if (threadIdx.x == 0) {
        float m = -3.4e38f;
        for (int i = 0; i < EE; ++i) m = fmaxf(m, lg[i]);
        float p[EE], s = 0.f;
        for (int i = 0; i < EE; ++i) { p[i] = __expf(lg[i] - m); s += p[i]; }
        for (int i = 0; i < EE; ++i) { p[i] /= s; cw[(size_t)bt * EE + i] = p[i]; }
        int idx[EE];
        for (int i = 0; i < EE; ++i) idx[i] = i;
        for (int i = 0; i < EE; ++i) {            // selection sort, descending
            int best = i;
            for (int j = i + 1; j < EE; ++j)
                if (p[idx[j]] > p[idx[best]]) best = j;
            int tmp = idx[i]; idx[i] = idx[best]; idx[best] = tmp;
        }
        for (int i = 0; i < EE; ++i) {
            topi[(size_t)bt * EE + i] = (float)idx[i];
            topp[(size_t)bt * EE + i] = p[idx[i]];  // sum==1 -> already normalized
        }
    }
}

// ---------------- swiglu: sw = b*sigmoid(b)*a ----------------
__global__ void swiglu_kernel(const float* __restrict__ ab, float* __restrict__ sw) {
    int idx = blockIdx.x * blockDim.x + threadIdx.x;   // 0 .. B*T*FF
    int m = idx >> 11;            // /FF
    int f = idx & (FF - 1);
    float a  = ab[(size_t)m * (2 * FF) + f];
    float bb = ab[(size_t)m * (2 * FF) + FF + f];
    sw[idx] = bb * (1.0f / (1.0f + __expf(-bb))) * a;
}

// ---------------- host orchestration ----------------
extern "C" void kernel_launch(void* const* d_in, const int* in_sizes, int n_in,
                              void* d_out, int out_size, void* d_ws, size_t ws_size,
                              hipStream_t stream) {
    const float* x     = (const float*)d_in[0];
    const float* g     = (const float*)d_in[1];
    const float* brms  = (const float*)d_in[2];
    const float* Wqkv  = (const float*)d_in[3];
    const float* bqkv  = (const float*)d_in[4];
    const float* alpha = (const float*)d_in[5];
    const float* Wproj = (const float*)d_in[6];
    const float* bproj = (const float*)d_in[7];
    const float* Wr    = (const float*)d_in[8];
    const float* br    = (const float*)d_in[9];
    const float* W1    = (const float*)d_in[10];
    const float* b1    = (const float*)d_in[11];
    const float* Wg    = (const float*)d_in[12];
    const float* bg    = (const float*)d_in[13];
    const float* W2    = (const float*)d_in[14];
    const float* b2    = (const float*)d_in[15];
    const float* W3    = (const float*)d_in[16];
    const float* b3    = (const float*)d_in[17];

    const size_t NBT = (size_t)BB * TT;                 // 1024
    const size_t NX  = NBT * CC;                        // 524288

    float* ws    = (float*)d_ws;
    float* x1    = ws;                                  // NX
    float* cw    = ws + NX;                             // NBT*E
    float* arena = ws + NX + NBT * EE;
    float* xn    = arena;                               // NX
    float* qkv   = arena + NX;                          // NBT*3C
    float* qh    = arena + NX * 4;                      // NBT*C each
    float* kh    = qh + NX;
    float* vh    = kh + NX;
    float* y     = arena;                               // reuse xn
    float* hbuf  = arena;                               // NBT*FF (reuse after proj)
    float* ab    = arena + NBT * FF;                    // NBT*2FF
    float* swb   = arena + NBT * FF * 3;                // NBT*FF
    float* s2    = arena + NBT * FF * 4;                // NBT*FF

    float* xout   = (float*)d_out;
    float* topi   = xout + NX;
    float* topp   = topi + NBT * TOPK;
    float* logits = topp + NBT * TOPK;

    // 1. RMSNorm
    rmsnorm_kernel<<<NBT, 256, 0, stream>>>(x, g, brms, xn);
    // 2. QKV GEMM: (1024,512)@(512,1536)
    gemm_wmma<0><<<dim3(3 * CC / BN, NBT / BM), 128, 0, stream>>>(
        xn, Wqkv, bqkv, nullptr, 0, qkv, NBT, 3 * CC, CC);
    // 3. QK-norm + RoPE + head split
    qk_rope_kernel<<<BB * TT * HH, DD, 0, stream>>>(qkv, alpha, qh, kh, vh);
    // 4. causal attention
    attention_kernel<<<BB * HH * TT, 128, 0, stream>>>(qh, kh, vh, y);
    // 5. proj + residual: x1 = y@Wproj + bproj + x
    gemm_wmma<1><<<dim3(CC / BN, NBT / BM), 128, 0, stream>>>(
        y, Wproj, bproj, x, 0, x1, NBT, CC, CC);
    // 6. router outputs (logits, top_i, top_p) + combine weights
    router_kernel<<<NBT, 32, 0, stream>>>(x1, Wr, br, logits, topi, topp, cw);
    // 7. seed output x with residual x1
    hipMemcpyAsync(xout, x1, NX * sizeof(float), hipMemcpyDeviceToDevice, stream);
    // 8. dense experts, weighted-accumulated into xout
    for (int e = 0; e < EE; ++e) {
        const float* W1e = W1 + (size_t)e * CC * FF;
        const float* b1e = b1 + (size_t)e * FF;
        const float* Wge = Wg + (size_t)e * FF * 2 * FF;
        const float* bge = bg + (size_t)e * 2 * FF;
        const float* W2e = W2 + (size_t)e * FF * FF;
        const float* b2e = b2 + (size_t)e * FF;
        const float* W3e = W3 + (size_t)e * FF * CC;
        const float* b3e = b3 + (size_t)e * CC;

        gemm_wmma<0><<<dim3(FF / BN, NBT / BM), 128, 0, stream>>>(
            x1, W1e, b1e, nullptr, 0, hbuf, NBT, FF, CC);
        gemm_wmma<0><<<dim3(2 * FF / BN, NBT / BM), 128, 0, stream>>>(
            hbuf, Wge, bge, nullptr, 0, ab, NBT, 2 * FF, FF);
        swiglu_kernel<<<(NBT * FF) / 256, 256, 0, stream>>>(ab, swb);
        gemm_wmma<0><<<dim3(FF / BN, NBT / BM), 128, 0, stream>>>(
            swb, W2e, b2e, nullptr, 0, s2, NBT, FF, FF);
        gemm_wmma<2><<<dim3(CC / BN, NBT / BM), 128, 0, stream>>>(
            s2, W3e, b3e, cw + e, EE, xout, NBT, CC, FF);
    }
}
